// VectorQuantization_54752243089898
// MI455X (gfx1250) — matrix operand order repaired and checked by reading, not compiled, not run on hardware
//
#include <hip/hip_runtime.h>

// CDNA5 / gfx1250: wave32, WMMA 16x16x32 bf16 -> f32 accumulate.
typedef __attribute__((ext_vector_type(16))) __bf16 v16bf;
typedef __attribute__((ext_vector_type(8)))  float  v8f;
typedef __attribute__((ext_vector_type(4)))  float  v4f;

#define NROWS       32768
#define KCB         8192
#define DDIM        512
#define ROWS_PER_WG 128          // 8 waves x 16 rows
#define GROUP_ROWS  128          // codebook rows staged per LDS group (8 col tiles)
#define NGROUPS     (KCB / GROUP_ROWS)   // 64
#define BSTRIDE     520          // padded bf16 row stride in LDS (bank-conflict break)
#define LOG2E       1.44269504088896f

__global__ __launch_bounds__(256) void vq_softmax_kernel(
    const float* __restrict__ X,      // [32768, 512]
    const float* __restrict__ CB,     // [8192, 512]
    float* __restrict__ OUT)          // [32768, 8192]
{
  extern __shared__ char smem[];
  __bf16* Bl   = (__bf16*)smem;                                  // 128*520 bf16 = 130 KiB
  float*  cbsq = (float*)(smem + GROUP_ROWS * BSTRIDE * 2);      // 128 floats: ||c||^2
  float*  ssp  = cbsq + GROUP_ROWS;                              // 256 floats: staging partials

  const int tid  = threadIdx.x;
  const int lane = tid & 31;
  const int wave = tid >> 5;
  const int half = lane >> 4;     // which 16-lane half
  const int l16  = lane & 15;

  const int rowbase = blockIdx.x * ROWS_PER_WG + wave * 16;

  // ---- Load this wave's A row tile (16 rows x 512 K) once, as bf16 fragments.
  // A-frag layout (16x32 bf16): lanes 0-15 hold K = 0..15; lanes 16-31
  // hold K = 16..31 of each 32-wide K chunk.
  v16bf a[16];
  {
    const float* xrow = X + (size_t)(rowbase + l16) * DDIM;
    #pragma unroll
    for (int d = 0; d < 16; ++d) {
      const int c0 = d * 32 + half * 16;
      v4f x0 = *(const v4f*)(xrow + c0);
      v4f x1 = *(const v4f*)(xrow + c0 + 4);
      v4f x2 = *(const v4f*)(xrow + c0 + 8);
      v4f x3 = *(const v4f*)(xrow + c0 + 12);
      v16bf t;
      t[0]  = (__bf16)x0[0]; t[1]  = (__bf16)x0[1]; t[2]  = (__bf16)x0[2]; t[3]  = (__bf16)x0[3];
      t[4]  = (__bf16)x1[0]; t[5]  = (__bf16)x1[1]; t[6]  = (__bf16)x1[2]; t[7]  = (__bf16)x1[3];
      t[8]  = (__bf16)x2[0]; t[9]  = (__bf16)x2[1]; t[10] = (__bf16)x2[2]; t[11] = (__bf16)x2[3];
      t[12] = (__bf16)x3[0]; t[13] = (__bf16)x3[1]; t[14] = (__bf16)x3[2]; t[15] = (__bf16)x3[3];
      a[d] = t;
    }
  }

  // Online-softmax state per lane: row = rowbase + v + 8*half (C-matrix layout).
  float m[8], s[8], rinv[8];
  #pragma unroll
  for (int v = 0; v < 8; ++v) { m[v] = -1e30f; s[v] = 0.0f; rinv[v] = 0.0f; }

  // Per-thread staging assignment (constant across groups).
  const int sr   = tid >> 1;              // 0..127 (group-local codebook row)
  const int scb0 = (tid & 1) * 256;       // half of the 512-wide row

  for (int pass = 0; pass < 2; ++pass) {
    for (int g = 0; g < NGROUPS; ++g) {
      __syncthreads();   // previous group's B fully consumed

      // ---- Cooperative stage: codebook rows [g*128, g*128+128) -> LDS as bf16,
      //      accumulating sum-of-squares (||c||^2) along the way.
      {
        const float* crow = CB + (size_t)(g * GROUP_ROWS + sr) * DDIM + scb0;
        __bf16* dst = Bl + sr * BSTRIDE + scb0;
        float acc = 0.0f;
        #pragma unroll 8
        for (int i = 0; i < 64; ++i) {
          v4f x = *(const v4f*)(crow + i * 4);
          acc += x[0]*x[0] + x[1]*x[1] + x[2]*x[2] + x[3]*x[3];
          dst[i*4+0] = (__bf16)x[0]; dst[i*4+1] = (__bf16)x[1];
          dst[i*4+2] = (__bf16)x[2]; dst[i*4+3] = (__bf16)x[3];
        }
        ssp[tid] = acc;
      }
      __syncthreads();
      if (tid < GROUP_ROWS) cbsq[tid] = ssp[tid * 2] + ssp[tid * 2 + 1];
      __syncthreads();

      // ---- Warm WGP$ with next group's staging lines while we compute this one.
      if (g + 1 < NGROUPS) {
        const float* nrow = CB + (size_t)((g + 1) * GROUP_ROWS + sr) * DDIM + scb0;
        #pragma unroll
        for (int i = 0; i < 8; ++i)
          __builtin_prefetch(nrow + i * 32, 0, 3);   // 8 x 128B cachelines
      }

      // ---- 8 column tiles per group, processed 4 at a time:
      //      4 independent WMMA accumulation chains to cover XDL latency.
      for (int tq = 0; tq < 2; ++tq) {
        const int t0 = tq * 4;
        v8f c0 = {}, c1 = {}, c2 = {}, c3 = {};
        // B-frag layout (32x16 bf16): lane holds 16 contiguous K of column l16;
        // lanes 0-15 -> K 0..15, lanes 16-31 -> K 16..31.
        const __bf16* bp = Bl + (t0 * 16 + l16) * BSTRIDE + half * 16;
        #pragma unroll
        for (int d = 0; d < 16; ++d) {
          v16bf fb0 = *(const v16bf*)(bp + 0 * 16 * BSTRIDE + d * 32);
          v16bf fb1 = *(const v16bf*)(bp + 1 * 16 * BSTRIDE + d * 32);
          v16bf fb2 = *(const v16bf*)(bp + 2 * 16 * BSTRIDE + d * 32);
          v16bf fb3 = *(const v16bf*)(bp + 3 * 16 * BSTRIDE + d * 32);
          c0 = __builtin_amdgcn_wmma_f32_16x16x32_bf16(false, a[d], false, fb0,
                                                       (short)0, c0, false, false);
          c1 = __builtin_amdgcn_wmma_f32_16x16x32_bf16(false, a[d], false, fb1,
                                                       (short)0, c1, false, false);
          c2 = __builtin_amdgcn_wmma_f32_16x16x32_bf16(false, a[d], false, fb2,
                                                       (short)0, c2, false, false);
          c3 = __builtin_amdgcn_wmma_f32_16x16x32_bf16(false, a[d], false, fb3,
                                                       (short)0, c3, false, false);
        }
        const float q0 = cbsq[(t0 + 0) * 16 + l16];
        const float q1 = cbsq[(t0 + 1) * 16 + l16];
        const float q2 = cbsq[(t0 + 2) * 16 + l16];
        const float q3 = cbsq[(t0 + 3) * 16 + l16];
        if (pass == 0) {
          #pragma unroll
          for (int v = 0; v < 8; ++v) {
            float L0 = (2.0f * c0[v] - q0) * LOG2E;   // logits in log2 units
            float L1 = (2.0f * c1[v] - q1) * LOG2E;
            float L2 = (2.0f * c2[v] - q2) * LOG2E;
            float L3 = (2.0f * c3[v] - q3) * LOG2E;
            float nm = fmaxf(fmaxf(m[v], fmaxf(L0, L1)), fmaxf(L2, L3));
            s[v] = s[v] * __builtin_amdgcn_exp2f(m[v] - nm)
                 + __builtin_amdgcn_exp2f(L0 - nm)
                 + __builtin_amdgcn_exp2f(L1 - nm)
                 + __builtin_amdgcn_exp2f(L2 - nm)
                 + __builtin_amdgcn_exp2f(L3 - nm);
            m[v] = nm;
          }
        } else {
          const int colb = (g * 8 + t0) * 16 + l16;
          #pragma unroll
          for (int v = 0; v < 8; ++v) {
            float L0 = (2.0f * c0[v] - q0) * LOG2E;
            float L1 = (2.0f * c1[v] - q1) * LOG2E;
            float L2 = (2.0f * c2[v] - q2) * LOG2E;
            float L3 = (2.0f * c3[v] - q3) * LOG2E;
            float* orow = OUT + (size_t)(rowbase + v + half * 8) * KCB + colb;
            orow[0]  = __builtin_amdgcn_exp2f(L0 - m[v]) * rinv[v];
            orow[16] = __builtin_amdgcn_exp2f(L1 - m[v]) * rinv[v];
            orow[32] = __builtin_amdgcn_exp2f(L2 - m[v]) * rinv[v];
            orow[48] = __builtin_amdgcn_exp2f(L3 - m[v]) * rinv[v];
          }
        }
      }
    }

    if (pass == 0) {
      // Merge (m,s) across the 16 lanes of each half (xor masks stay within a half).
      #pragma unroll
      for (int v = 0; v < 8; ++v) {
        float mv = m[v], sv = s[v];
        #pragma unroll
        for (int mask = 1; mask < 16; mask <<= 1) {
          float om = __shfl_xor(mv, mask, 32);
          float os = __shfl_xor(sv, mask, 32);
          float nm = fmaxf(mv, om);
          sv = sv * __builtin_amdgcn_exp2f(mv - nm)
             + os * __builtin_amdgcn_exp2f(om - nm);
          mv = nm;
        }
        m[v] = mv;
        rinv[v] = 1.0f / sv;
      }
    }
  }
}

extern "C" void kernel_launch(void* const* d_in, const int* in_sizes, int n_in,
                              void* d_out, int out_size, void* d_ws, size_t ws_size,
                              hipStream_t stream) {
  (void)in_sizes; (void)n_in; (void)d_ws; (void)ws_size; (void)out_size;
  const float* X  = (const float*)d_in[0];   // inputs  [32768*512]
  const float* CB = (const float*)d_in[1];   // codebook [8192*512]
  float* OUT = (float*)d_out;                // [32768*8192]

  const size_t shmem = (size_t)GROUP_ROWS * BSTRIDE * 2   // B tile (bf16)
                     + GROUP_ROWS * sizeof(float)         // cbsq
                     + 256 * sizeof(float);               // staging partials
  dim3 grid(NROWS / ROWS_PER_WG);   // 256
  dim3 block(256);                  // 8 wave32
  hipLaunchKernelGGL(vq_softmax_kernel, grid, block, shmem, stream, X, CB, OUT);
}
